// QuadrupletInteraction_78408922956498
// MI455X (gfx1250) — compile-verified, matrix operand order checked by reading, not run on hardware
//
#include <hip/hip_runtime.h>
#include <hip/hip_bf16.h>

typedef _Float16 h16;
typedef __attribute__((ext_vector_type(16))) _Float16 v16h;
typedef __attribute__((ext_vector_type(8)))  _Float16 v8h;
typedef __attribute__((ext_vector_type(8)))  float    v8f;

#define DE   128
#define DQd  64
#define DBd  64
#define DRBF 16
#define DCBF 16
#define DSBF 32
#define NSPH 32
#define KMAXP 16

// ---- WMMA helpers (CDNA5 wave32, V_WMMA_F32_16X16X32_F16) --------------------

// half-index -> K mapping shared by the 16-bit A and B operand layouts:
// lanes 0-15: K = {0..7, 16..23}; lanes 16-31: K = {8..15, 24..31}
__device__ __forceinline__ int kmap(int lane, int j) {
  int klo = (lane & 16) ? 8 : 0;
  return (j < 8) ? (klo + j) : (8 + klo + j);
}

__device__ __forceinline__ float sact(float x) {  // ScaledSiLU
  return x * (1.0f / (1.0f + __expf(-x))) * (1.0f / 0.6f);
}

__device__ __forceinline__ v8f wmma16(v16h a, v16h b, v8f c) {
  return __builtin_amdgcn_wmma_f32_16x16x32_f16(false, a, false, b, (short)0, c,
                                                false, false);
}

// A fragment from CONTIGUOUS fp32 row (stride 1 along K). klimit multiple of 16.
__device__ __forceinline__ v16h afrag_f32_c(const float* base, int lane,
                                            int kbase, int klimit) {
  int klo = (lane & 16) ? 8 : 0;
  float4 f0 = {0.f, 0.f, 0.f, 0.f}, f1 = f0, f2 = f0, f3 = f0;
  if (kbase + 16 <= klimit) {
    const float4* p = (const float4*)(base + kbase + klo);
    f0 = p[0]; f1 = p[1];
  }
  if (kbase + 32 <= klimit) {
    const float4* p = (const float4*)(base + kbase + 16 + klo);
    f2 = p[0]; f3 = p[1];
  }
  v16h a;
  a[0] = (h16)f0.x; a[1] = (h16)f0.y; a[2]  = (h16)f0.z; a[3]  = (h16)f0.w;
  a[4] = (h16)f1.x; a[5] = (h16)f1.y; a[6]  = (h16)f1.z; a[7]  = (h16)f1.w;
  a[8] = (h16)f2.x; a[9] = (h16)f2.y; a[10] = (h16)f2.z; a[11] = (h16)f2.w;
  a[12] = (h16)f3.x; a[13] = (h16)f3.y; a[14] = (h16)f3.z; a[15] = (h16)f3.w;
  return a;
}

// A fragment from contiguous f16 row (e.g. LDS): two aligned 16B loads.
__device__ __forceinline__ v16h afrag_f16_c(const h16* base, int lane, int kbase) {
  int klo = (lane & 16) ? 8 : 0;
  v8h lo = *(const v8h*)(base + kbase + klo);
  v8h hi = *(const v8h*)(base + kbase + 16 + klo);
  v16h a;
#pragma unroll
  for (int j = 0; j < 8; ++j) { a[j] = lo[j]; a[j + 8] = hi[j]; }
  return a;
}

// B fragment from pre-swizzled packed weights: one aligned 32B load per lane.
__device__ __forceinline__ v16h bfrag_packed(const h16* pb, int frag, int lane) {
  return *(const v16h*)(pb + ((size_t)frag * 32 + lane) * 16);
}

// A fragment of sph^T: element K=k at base[k*NSPH]; only k<16 valid.
__device__ __forceinline__ v16h afrag_sph(const float* base, int lane) {
  int klo = (lane & 16) ? 8 : 0;
  v16h a;
#pragma unroll
  for (int j = 0; j < 8; ++j) {
    a[j] = (h16)base[(size_t)(klo + j) * NSPH];
    a[j + 8] = (h16)0.0f;
  }
  return a;
}

// B fragment of packed m2 (k<16 valid): one aligned 16B load.
__device__ __forceinline__ v16h bfrag_m2(const h16* pm2e, int nt, int lane) {
  v8h lo = *(const v8h*)(pm2e + nt * 256 + lane * 8);
  v16h b;
#pragma unroll
  for (int j = 0; j < 8; ++j) { b[j] = lo[j]; b[j + 8] = (h16)0.0f; }
  return b;
}

// ---- Kernel 1: weights -> f16, pre-swizzled into B-fragment order -----------
// packed[((ks*Ntiles + nt)*32 + lane)*16 + j] = W[(ks*32 + kmap(lane,j))*N + nt*16 + lane%16]

__device__ __forceinline__ void pack_decode(int g, int Ntiles, int* k, int* n) {
  int j = g & 15, lane = (g >> 4) & 31, rest = g >> 9;
  int nt = rest % Ntiles, ks = rest / Ntiles;
  *k = ks * 32 + kmap(lane, j);
  *n = nt * 16 + (lane & 15);
}

__global__ void k_convert(const float* Wdb, const float* Wrbf, const float* Wdown,
                          const float* Wcbf, const float* Wbilin, const float* Wca,
                          const float* Wac,
                          h16* wdb, h16* wrbf, h16* wdown, h16* wcbf,
                          h16* wb, h16* wca, h16* wac) {
  int g = blockIdx.x * 256 + threadIdx.x;
  int k, n;
  if (g < 16384) {  // W_db 128x128
    pack_decode(g, 8, &k, &n);
    wdb[g] = (h16)Wdb[k * 128 + n]; return;
  }
  g -= 16384;
  if (g < 4096) {   // W_rbf 16x128 padded to K=32
    pack_decode(g, 8, &k, &n);
    wrbf[g] = (h16)((k < DRBF) ? Wrbf[k * 128 + n] : 0.0f); return;
  }
  g -= 4096;
  if (g < 8192) {   // W_down 128x64
    pack_decode(g, 4, &k, &n);
    wdown[g] = (h16)Wdown[k * 64 + n]; return;
  }
  g -= 8192;
  if (g < 2048) {   // W_cbf 16x64 padded to K=32
    pack_decode(g, 4, &k, &n);
    wcbf[g] = (h16)((k < DCBF) ? Wcbf[k * 64 + n] : 0.0f); return;
  }
  g -= 2048;
  if (g < 8192) {   // W_up_ca 64x128
    pack_decode(g, 8, &k, &n);
    wca[g] = (h16)Wca[k * 128 + n]; return;
  }
  g -= 8192;
  if (g < 8192) {   // W_up_ac 64x128
    pack_decode(g, 8, &k, &n);
    wac[g] = (h16)Wac[k * 128 + n]; return;
  }
  g -= 8192;
  if (g < 131072) { // W_bilin: logical WB[k=i*64+q][b], K=2048, N=64
    pack_decode(g, 4, &k, &n);
    int i = k >> 6, q = k & 63;
    wb[g] = (h16)Wbilin[((size_t)q * 32 + i) * 64 + n]; return;
  }
}

// ---- Kernel 2: edge stage: act(m@W_db) * (rbf@W_rbf) * s_rbf -> act(@W_down) --

__global__ void k_edge(const float* m, const float* rbf, const float* s_rbf,
                       const h16* wdb, const h16* wrbf, const h16* wdown,
                       h16* x_edge) {
  __shared__ __attribute__((aligned(32))) h16 ts[16 * DE];  // 4 KB transpose buf
  int lane = threadIdx.x & 31;
  int wave = threadIdx.x >> 5;  // 0..3
  int l15 = lane & 15;
  int hi8 = (lane & 16) ? 8 : 0;
  int e0 = blockIdx.x * 16;
  float srbf = *s_rbf;
  const float* mrow = m + (size_t)(e0 + l15) * DE;
  const float* rrow = rbf + (size_t)(e0 + l15) * DRBF;
  v16h am[4];
#pragma unroll
  for (int ks = 0; ks < 4; ++ks) am[ks] = afrag_f32_c(mrow, lane, ks * 32, DE);
  v16h ar = afrag_f32_c(rrow, lane, 0, DRBF);
#pragma unroll
  for (int tt = 0; tt < 2; ++tt) {
    int nt = wave * 2 + tt;
    int nb = nt * 16;
    v8f cdb = {};
#pragma unroll
    for (int ks = 0; ks < 4; ++ks)
      cdb = wmma16(am[ks], bfrag_packed(wdb, ks * 8 + nt, lane), cdb);
    v8f crb = {};
    crb = wmma16(ar, bfrag_packed(wrbf, nt, lane), crb);
#pragma unroll
    for (int r = 0; r < 8; ++r) {
      float t = sact(cdb[r]) * crb[r] * srbf;
      ts[(r + hi8) * DE + nb + l15] = (h16)t;
    }
  }
  __syncthreads();
  int nt2 = wave;
  v8f c = {};
#pragma unroll
  for (int ks = 0; ks < 4; ++ks) {
    v16h a = afrag_f16_c(&ts[l15 * DE], lane, ks * 32);
    c = wmma16(a, bfrag_packed(wdown, ks * 4 + nt2, lane), c);
  }
#pragma unroll
  for (int r = 0; r < 8; ++r) {
    int row = e0 + r + hi8;
    x_edge[(size_t)row * DQd + nt2 * 16 + l15] = (h16)sact(c[r]);
  }
}

// ---- Kernel 3: y[i] = x_edge[id_intm[i]] * (cbf@W_cbf) * s_cbf ---------------

__global__ void k_cbf(const float* cbf, const float* s_cbf, const h16* wcbf,
                      const h16* x_edge, const int* id_intm, h16* y) {
  int lane = threadIdx.x & 31;
  int wave = threadIdx.x >> 5;  // 0..1
  int l15 = lane & 15, hi8 = (lane & 16) ? 8 : 0;
  int i0 = blockIdx.x * 16;
  float sc = *s_cbf;
  v16h a = afrag_f32_c(cbf + (size_t)(i0 + l15) * DCBF, lane, 0, DCBF);
#pragma unroll
  for (int tt = 0; tt < 2; ++tt) {
    int nt = wave * 2 + tt;
    int nb = nt * 16;
    v8f c = {};
    c = wmma16(a, bfrag_packed(wcbf, nt, lane), c);
#pragma unroll
    for (int r = 0; r < 8; ++r) {
      int row = i0 + r + hi8;
      int src = id_intm[row];
      float xe = (float)x_edge[(size_t)src * DQd + nb + l15];
      y[(size_t)row * DQd + nb + l15] = (h16)(c[r] * xe * sc);
    }
  }
}

// ---- Kernel 4/5: zero packed m2, then scatter quads directly into ------------
// B-fragment order: pm2[e*1024 + nt*256 + lane*8 + (k&7)], lane = (k>=8?16:0)+col%16

__global__ void k_zero(unsigned int* p, long long n) {
  long long g = (long long)blockIdx.x * blockDim.x + threadIdx.x;
  if (g < n) p[g] = 0u;
}

__global__ void k_scatter(const h16* y, const int* reduce_ca, const int* kidx,
                          const int* expand_abd, h16* pm2, int Q) {
  int g = blockIdx.x * 256 + threadIdx.x;
  int q = g >> 4, c4 = g & 15;  // 4 columns per thread
  if (q >= Q) return;
  int e = reduce_ca[q], kk = kidx[q], aq = expand_abd[q];
  ushort4 v = *(const ushort4*)(y + (size_t)aq * DQd + c4 * 4);
  int nt = c4 >> 2;
  int lbase = ((kk >= 8) ? 16 : 0) + (c4 & 3) * 4;
  int j = kk & 7;
  unsigned short* du = (unsigned short*)(pm2 + (size_t)e * 1024 + nt * 256);
  du[(lbase + 0) * 8 + j] = v.x;
  du[(lbase + 1) * 8 + j] = v.y;
  du[(lbase + 2) * 8 + j] = v.z;
  du[(lbase + 3) * 8 + j] = v.w;
}

// ---- Kernel 6: per-edge sum_k / rw, then X(16x2048) @ WB(2048x64) -----------

__global__ void k_bilinear(const float* sph, const float* sbf, const float* s_sbf,
                           const h16* pm2, const h16* wb, float* xbil) {
  __shared__ __attribute__((aligned(32))) h16 Xs[16 * 2048];  // 64 KB
  int lane = threadIdx.x & 31;
  int wave = threadIdx.x >> 5;  // 0..7
  int l15 = lane & 15, hi8 = (lane & 16) ? 8 : 0;
  int e0 = blockIdx.x * 16;
#pragma unroll 1
  for (int w2 = 0; w2 < 2; ++w2) {
    int eloc = wave * 2 + w2;
    int e = e0 + eloc;
    // sum_k (32x64) = sph^T (32x16, K padded to 32) @ m2[e] (16x64, padded)
    const float* sphe = sph + (size_t)e * KMAXP * NSPH;
    v16h as0 = afrag_sph(sphe + (0 + l15), lane);
    v16h as1 = afrag_sph(sphe + (16 + l15), lane);
    const h16* pm2e = pm2 + (size_t)e * 1024;
    v8f S0[4], S1[4];
#pragma unroll
    for (int nt = 0; nt < 4; ++nt) {
      v16h b = bfrag_m2(pm2e, nt, lane);
      v8f z = {};
      S0[nt] = wmma16(as0, b, z);
      S1[nt] = wmma16(as1, b, z);
    }
    // rw (32x64) = sbf_W1[e] (32x32) @ sum_k; C->B conversion is lane-local
    v16h aw0 = afrag_f32_c(sbf + ((size_t)e * DSBF + 0 + l15) * NSPH, lane, 0, NSPH);
    v16h aw1 = afrag_f32_c(sbf + ((size_t)e * DSBF + 16 + l15) * NSPH, lane, 0, NSPH);
#pragma unroll
    for (int nt = 0; nt < 4; ++nt) {
      v16h b;
#pragma unroll
      for (int j = 0; j < 8; ++j)  b[j] = (h16)S0[nt][j];
#pragma unroll
      for (int j = 8; j < 16; ++j) b[j] = (h16)S1[nt][j - 8];
      v8f z = {};
      v8f R0 = wmma16(aw0, b, z);
      v8f R1 = wmma16(aw1, b, z);
#pragma unroll
      for (int r = 0; r < 8; ++r) {
        int q = nt * 16 + l15;
        Xs[eloc * 2048 + (r + hi8) * 64 + q]      = (h16)R0[r];
        Xs[eloc * 2048 + (16 + r + hi8) * 64 + q] = (h16)R1[r];
      }
    }
  }
  __syncthreads();
  // X (16 x 2048) @ WB (2048 x 64): 4 N-tiles x 2 K-halves over 8 waves
  int nt = wave & 3, kh = wave >> 2;
  v8f C = {};
#pragma unroll 1
  for (int s = 0; s < 32; ++s) {
    int kstep = kh * 32 + s;
    v16h a = afrag_f16_c(&Xs[l15 * 2048], lane, kstep * 32);
    v16h b = bfrag_packed(wb, kstep * 4 + nt, lane);
    C = wmma16(a, b, C);
  }
  __syncthreads();
  float* red = (float*)Xs;  // reuse LDS for the K-half reduction (4 KB)
  if (kh == 1) {
#pragma unroll
    for (int r = 0; r < 8; ++r) red[nt * 256 + r * 32 + lane] = C[r];
  }
  __syncthreads();
  if (kh == 0) {
    float ss = *s_sbf;
#pragma unroll
    for (int r = 0; r < 8; ++r) {
      float v = (C[r] + red[nt * 256 + r * 32 + lane]) * ss;
      int row = e0 + r + hi8;
      xbil[(size_t)row * DBd + nt * 16 + l15] = v;
    }
  }
}

// ---- Kernel 7: out = (act(x@Wca) + act(x[id_swap]@Wac)) / sqrt(2) -----------

__global__ void k_up(const float* xbil, const int* id_swap, const h16* wca,
                     const h16* wac, float* out) {
  int lane = threadIdx.x & 31;
  int wave = threadIdx.x >> 5;  // 0..3
  int l15 = lane & 15, hi8 = (lane & 16) ? 8 : 0;
  int e0 = blockIdx.x * 16;
  int row = e0 + l15;
  int swrow = id_swap[row];
  const float* xr = xbil + (size_t)row * DBd;
  const float* xs = xbil + (size_t)swrow * DBd;
  v16h ar0 = afrag_f32_c(xr, lane, 0, DBd);
  v16h ar1 = afrag_f32_c(xr, lane, 32, DBd);
  v16h as0 = afrag_f32_c(xs, lane, 0, DBd);
  v16h as1 = afrag_f32_c(xs, lane, 32, DBd);
#pragma unroll
  for (int tt = 0; tt < 2; ++tt) {
    int nt = wave * 2 + tt;
    int nb = nt * 16;
    v8f cca = {}, cac = {};
    cca = wmma16(ar0, bfrag_packed(wca, 0 * 8 + nt, lane), cca);
    cca = wmma16(ar1, bfrag_packed(wca, 1 * 8 + nt, lane), cca);
    cac = wmma16(as0, bfrag_packed(wac, 0 * 8 + nt, lane), cac);
    cac = wmma16(as1, bfrag_packed(wac, 1 * 8 + nt, lane), cac);
#pragma unroll
    for (int r = 0; r < 8; ++r) {
      int rr = e0 + r + hi8;
      out[(size_t)rr * DE + nb + l15] =
          (sact(cca[r]) + sact(cac[r])) * 0.70710678118f;
    }
  }
}

// ---- launch -----------------------------------------------------------------

extern "C" void kernel_launch(void* const* d_in, const int* in_sizes, int n_in,
                              void* d_out, int out_size, void* d_ws, size_t ws_size,
                              hipStream_t stream) {
  (void)n_in; (void)out_size; (void)ws_size;
  const float* m       = (const float*)d_in[0];
  const float* rbf     = (const float*)d_in[1];
  const float* cbf     = (const float*)d_in[2];
  const float* sbf_W1  = (const float*)d_in[3];
  const float* sph     = (const float*)d_in[4];
  const int*   Kidx4   = (const int*)d_in[5];
  const int*   id4_red = (const int*)d_in[6];
  const int*   id_swap = (const int*)d_in[7];
  const int*   id_intm = (const int*)d_in[8];
  const int*   id_abd  = (const int*)d_in[9];
  const float* W_db    = (const float*)d_in[10];
  const float* W_rbf   = (const float*)d_in[11];
  const float* W_down  = (const float*)d_in[12];
  const float* W_cbf   = (const float*)d_in[13];
  const float* W_bilin = (const float*)d_in[14];
  const float* W_ca    = (const float*)d_in[15];
  const float* W_ac    = (const float*)d_in[16];
  const float* s_rbf   = (const float*)d_in[17];
  const float* s_cbf   = (const float*)d_in[18];
  const float* s_sbf   = (const float*)d_in[19];

  int E = in_sizes[0] / DE;     // 30000
  int I = in_sizes[2] / DCBF;   // 180000
  int Q = in_sizes[9];          // 360000

  char* w = (char*)d_ws;
  size_t o = 0;
  h16* wdb    = (h16*)(w + o); o += 16384 * 2;
  h16* wrbf   = (h16*)(w + o); o += 4096 * 2;
  h16* wdown  = (h16*)(w + o); o += 8192 * 2;
  h16* wcbf   = (h16*)(w + o); o += 2048 * 2;
  h16* wca    = (h16*)(w + o); o += 8192 * 2;
  h16* wac    = (h16*)(w + o); o += 8192 * 2;
  h16* wb     = (h16*)(w + o); o += 131072 * 2;
  h16* x_edge = (h16*)(w + o); o += (size_t)E * DQd * 2;
  h16* y      = (h16*)(w + o); o += (size_t)I * DQd * 2;
  h16* pm2    = (h16*)(w + o); o += (size_t)E * KMAXP * DQd * 2;  // packed frag order
  float* xbil = (float*)(w + o); o += (size_t)E * DBd * 4;        // ~96.4 MB total

  k_convert<<<(178176 + 255) / 256, 256, 0, stream>>>(
      W_db, W_rbf, W_down, W_cbf, W_bilin, W_ca, W_ac,
      wdb, wrbf, wdown, wcbf, wb, wca, wac);
  k_edge<<<E / 16, 128, 0, stream>>>(m, rbf, s_rbf, wdb, wrbf, wdown, x_edge);
  k_cbf<<<I / 16, 64, 0, stream>>>(cbf, s_cbf, wcbf, x_edge, id_intm, y);
  long long nz = (long long)E * KMAXP * DQd / 2;  // uint32 count
  k_zero<<<(unsigned)((nz + 255) / 256), 256, 0, stream>>>((unsigned int*)pm2, nz);
  k_scatter<<<((size_t)Q * 16 + 255) / 256, 256, 0, stream>>>(y, id4_red, Kidx4,
                                                              id_abd, pm2, Q);
  k_bilinear<<<E / 16, 256, 0, stream>>>(sph, sbf_W1, s_sbf, pm2, wb, xbil);
  k_up<<<E / 16, 128, 0, stream>>>(xbil, id_swap, wca, wac, (float*)d_out);
}
